// VITS_42752104464929
// MI455X (gfx1250) — matrix operand order repaired and checked by reading, not compiled, not run on hardware
//
#include <hip/hip_runtime.h>
#include <hip/hip_bf16.h>

// Problem constants (match reference setup_inputs)
#define BB 8
#define DD 192
#define TY 8192
#define TX 1024
#define K2 (2 * DD)          // fused contraction dim = 384
#define KSTEPS (K2 / 32)     // 12 WMMA K-slabs
#define LOG2PI_F 1.8378770664093453f

typedef __attribute__((ext_vector_type(16))) __bf16 v16bf;
typedef __attribute__((ext_vector_type(8)))  __bf16 v8bf;
typedef __attribute__((ext_vector_type(8)))  float  v8f;

union ABFrag { v16bf v; v8bf h[2]; };

__device__ __forceinline__ unsigned short f2bf(float f) {
    unsigned u = __builtin_bit_cast(unsigned, f);
    u += 0x7FFFu + ((u >> 16) & 1u);          // round-to-nearest-even
    return (unsigned short)(u >> 16);
}

// ---------------------------------------------------------------------------
// Prep B-side operand U[b][y][k] (bf16): k<D -> -0.5*z^2 ; k>=D -> z
// LDS 32x32 transpose so both global read (over y) and write (over d) coalesce.
// ---------------------------------------------------------------------------
__global__ __launch_bounds__(1024) void prep_u(const float* __restrict__ z_p,
                                               unsigned short* __restrict__ U) {
    __shared__ float tile[32][33];
    const int b  = blockIdx.z;
    const int d0 = blockIdx.y * 32;
    const int y0 = blockIdx.x * 32;
    const int tx = threadIdx.x, ty = threadIdx.y;
    tile[ty][tx] = z_p[((size_t)(b * DD + d0 + ty)) * TY + y0 + tx];
    __syncthreads();
    const int d = d0 + tx, y = y0 + ty;
    const float z = tile[tx][ty];
    unsigned short* row = U + ((size_t)(b * TY) + y) * K2;
    row[d]      = f2bf(-0.5f * z * z);
    row[DD + d] = f2bf(z);
}

// ---------------------------------------------------------------------------
// Prep A-side operand Vm[b][x][k] (bf16): k<D -> s ; k>=D -> m*s
// and per-column constant c[b][x] = nc1 + nc4.
// ---------------------------------------------------------------------------
__global__ __launch_bounds__(256) void prep_v(const float* __restrict__ m_p,
                                              const float* __restrict__ logs_p,
                                              unsigned short* __restrict__ Vm,
                                              float* __restrict__ cArr) {
    __shared__ float red[256];
    const int b = blockIdx.x / TX;
    const int x = blockIdx.x % TX;
    const int d = threadIdx.x;
    float partial = 0.f;
    if (d < DD) {
        float lg = logs_p[((size_t)(b * DD + d)) * TX + x];
        lg = fminf(10.f, fmaxf(-10.f, lg));
        const float m = m_p[((size_t)(b * DD + d)) * TX + x];
        const float s = expf(-2.f * lg);
        unsigned short* row = Vm + ((size_t)(b * TX) + x) * K2;
        row[d]      = f2bf(s);
        row[DD + d] = f2bf(m * s);
        partial = -0.5f * LOG2PI_F - lg - 0.5f * m * m * s;
    }
    red[d] = partial;
    __syncthreads();
    for (int s2 = 128; s2 > 0; s2 >>= 1) {
        if (d < s2) red[d] += red[d + s2];
        __syncthreads();
    }
    if (d == 0) cArr[b * TX + x] = red[0];
}

// ---------------------------------------------------------------------------
// WMMA GEMM: negT[b][x][y] = sum_k Vm[b][x][k] * U[b][y][k] + c[b][x]
// A = Vm (M=x, 16 rows), B = U^T (N=y), K=384, bf16 in, f32 accum.
// One wave -> 16x64 output strip (4 N-tiles). Double-buffered fragments:
// slab k+1 A/B loads are issued before the four WMMAs of slab k, so the
// s_wait for the next slab lands behind WMMA issue instead of stalling.
// ---------------------------------------------------------------------------
__global__ __launch_bounds__(256) void gemm_negcent(const unsigned short* __restrict__ Vm,
                                                    const unsigned short* __restrict__ U,
                                                    const float* __restrict__ cArr,
                                                    float* __restrict__ negT) {
    const int wid  = (blockIdx.x << 3) + (threadIdx.x >> 5);
    const int lane = threadIdx.x & 31;
    const int per_b = (TX / 16) * (TY / 64);               // 64 * 128 = 8192
    if (wid >= BB * per_b) return;
    const int b  = wid / per_b;
    const int r  = wid % per_b;
    const int mt = r / (TY / 64);
    const int ng = r % (TY / 64);
    const int x0 = mt * 16, y0 = ng * 64;
    const int lm = lane & 15;
    const int hs = lane & 16;

    // A fragment (16-bit A 16x32 layout): lane<16 holds K {0..7,16..23} of row M=lm,
    // lane>=16 holds K {8..15,24..31}. Row-major Vm[x][k] -> two 16B loads per slab.
    const unsigned short* Arow = Vm + ((size_t)(b * TX + x0 + lm) * K2 + (hs ? 8 : 0));
    // B fragment (16-bit B 32x16 layout): lane holds column N=y, K {0..7,16..23} (lane<16) or {8..15,24..31}.
    const unsigned short* Brow0 = U + ((size_t)(b * TY + y0 + lm) * K2 + (hs ? 8 : 0));

    const v8f vzero = {0.f, 0.f, 0.f, 0.f, 0.f, 0.f, 0.f, 0.f};
    v8f acc[4];
#pragma unroll
    for (int i = 0; i < 4; ++i) acc[i] = vzero;

    ABFrag a[2];
    ABFrag bf[2][4];

    // Preload slab 0
    a[0].h[0] = *(const v8bf*)(Arow);
    a[0].h[1] = *(const v8bf*)(Arow + 16);
#pragma unroll
    for (int nt = 0; nt < 4; ++nt) {
        const unsigned short* br = Brow0 + (size_t)nt * 16 * K2;
        bf[0][nt].h[0] = *(const v8bf*)(br);
        bf[0][nt].h[1] = *(const v8bf*)(br + 16);
    }

#pragma unroll
    for (int step = 0; step < KSTEPS; ++step) {
        const int cur = step & 1, nxt = cur ^ 1;
        if (step + 1 < KSTEPS) {
            const int kn = (step + 1) * 32;
            a[nxt].h[0] = *(const v8bf*)(Arow + kn);
            a[nxt].h[1] = *(const v8bf*)(Arow + kn + 16);
            if (step + 2 < KSTEPS) {
                __builtin_prefetch(Arow + kn + 32, 0, 1);   // global_prefetch_b8
                __builtin_prefetch(Brow0 + kn + 32, 0, 1);
            }
#pragma unroll
            for (int nt = 0; nt < 4; ++nt) {
                const unsigned short* br = Brow0 + (size_t)nt * 16 * K2 + kn;
                bf[nxt][nt].h[0] = *(const v8bf*)(br);
                bf[nxt][nt].h[1] = *(const v8bf*)(br + 16);
            }
        }
#pragma unroll
        for (int nt = 0; nt < 4; ++nt) {
            acc[nt] = __builtin_amdgcn_wmma_f32_16x16x32_bf16(
                false, a[cur].v, false, bf[cur][nt].v, (short)0, acc[nt], false, false);
        }
    }

    // D tile: VGPR rr, lane -> (M = rr + (lane>=16 ? 8:0), N = lane&15).
    const int xoff = hs ? 8 : 0;
    const float* cb = cArr + b * TX + x0 + xoff;
#pragma unroll
    for (int nt = 0; nt < 4; ++nt) {
#pragma unroll
        for (int rr = 0; rr < 8; ++rr) {
            const int x = x0 + rr + xoff;
            negT[((size_t)(b * TX) + x) * TY + (y0 + nt * 16 + lm)] = acc[nt][rr] + cb[rr];
        }
    }
}

// ---------------------------------------------------------------------------
// MAS forward DP: 1 workgroup per batch; DP row (8192 f32) lives in LDS.
// direction bits packed 32/word via wave32 ballot -> dir[b][j][256] u32.
// ---------------------------------------------------------------------------
__global__ __launch_bounds__(1024) void dp_fwd(const float* __restrict__ negT,
                                               const int* __restrict__ ylen_,
                                               unsigned* __restrict__ dir) {
    __shared__ float v[TY];
    const int b = blockIdx.x;
    const int t = threadIdx.x;
    const int ylen = ylen_[b];
    const float NEG_INF = -__builtin_inff();
#pragma unroll
    for (int i = 0; i < TY / 1024; ++i) v[i * 1024 + t] = 0.f;
    __syncthreads();

    for (int j = 0; j < TX; ++j) {
        const float* col = negT + ((size_t)(b * TX + j)) * TY;   // contiguous in y
        float nv[TY / 1024];
        bool  st[TY / 1024];
#pragma unroll
        for (int i = 0; i < TY / 1024; ++i) {
            const int y = i * 1024 + t;
            const float cur  = v[y];
            const float prev = (y == 0) ? NEG_INF : v[y - 1];
            const bool stay  = cur >= prev;
            const float vm   = stay ? cur : prev;
            const bool valid = (y <= j) && (y < ylen);
            nv[i] = valid ? vm + col[y] : NEG_INF;
            st[i] = stay;
        }
#pragma unroll
        for (int i = 0; i < TY / 1024; ++i) {
            const unsigned long long m = __ballot((int)st[i]);   // wave32: low 32 bits
            if ((t & 31) == 0)
                dir[((size_t)(b * TX + j)) * (TY / 32) + (i * 32 + (t >> 5))] = (unsigned)m;
        }
        __syncthreads();
#pragma unroll
        for (int i = 0; i < TY / 1024; ++i) v[i * 1024 + t] = nv[i];
        __syncthreads();
    }
}

// ---------------------------------------------------------------------------
// Backtrack: one thread per batch, 1024 sequential steps.
// ---------------------------------------------------------------------------
__global__ void backtrack(const unsigned* __restrict__ dir,
                          const int* __restrict__ ylen_,
                          int* __restrict__ idxA) {
    const int b = threadIdx.x;
    if (b >= BB) return;
    int idx = ylen_[b] - 1;
    for (int j = TX - 1; j >= 0; --j) {
        idxA[b * TX + j] = idx;
        const unsigned w = dir[((size_t)(b * TX + j)) * (TY / 32) + (idx >> 5)];
        const bool d = (w >> (idx & 31)) & 1u;
        if (j > 0 && !d && idx > 0) idx--;
    }
}

// attn[b][y][x] = (idxA[b][x] == y)
__global__ __launch_bounds__(256) void write_attn(const int* __restrict__ idxA,
                                                  float* __restrict__ out) {
    const size_t id = (size_t)blockIdx.x * 256 + threadIdx.x;
    const int x = (int)(id % TX);
    const size_t t = id / TX;
    const int y = (int)(t % TY);
    const int b = (int)(t / TY);
    out[id] = (idxA[b * TX + x] == y) ? 1.0f : 0.0f;
}

// w: exactly one path entry per source column -> all ones
__global__ void write_w(float* __restrict__ w) {
    const int id = blockIdx.x * 256 + threadIdx.x;
    if (id < BB * TX) w[id] = 1.0f;
}

// m_p_exp / logs_p_exp: idxA is non-decreasing in x, so per (b,y) the mapped
// columns form a contiguous range; binary-search it and sum. Deterministic.
__global__ __launch_bounds__(256) void expand(const int* __restrict__ idxA,
                                              const float* __restrict__ m_p,
                                              const float* __restrict__ logs_p,
                                              float* __restrict__ mo,
                                              float* __restrict__ lo_) {
    const size_t id = (size_t)blockIdx.x * 256 + threadIdx.x;
    const int y = (int)(id % TY);
    const size_t t = id / TY;
    const int d = (int)(t % DD);
    const int b = (int)(t / DD);
    const int* ia = idxA + b * TX;
    int lo = 0, hi = TX;
    while (lo < hi) { const int mid = (lo + hi) >> 1; if (ia[mid] < y) lo = mid + 1; else hi = mid; }
    int lo2 = lo, hi2 = TX;
    while (lo2 < hi2) { const int mid = (lo2 + hi2) >> 1; if (ia[mid] < y + 1) lo2 = mid + 1; else hi2 = mid; }
    float sm = 0.f, sl = 0.f;
    const size_t base = ((size_t)(b * DD + d)) * TX;
    for (int x = lo; x < lo2; ++x) { sm += m_p[base + x]; sl += logs_p[base + x]; }
    mo[id] = sm;
    lo_[id] = sl;
}

extern "C" void kernel_launch(void* const* d_in, const int* in_sizes, int n_in,
                              void* d_out, int out_size, void* d_ws, size_t ws_size,
                              hipStream_t stream) {
    (void)in_sizes; (void)n_in; (void)out_size; (void)ws_size;
    const float* z_p    = (const float*)d_in[0];
    const float* m_p    = (const float*)d_in[1];
    const float* logs_p = (const float*)d_in[2];
    // d_in[3] = x_lengths (always TX in this setup, x_mask all-true)
    const int*   ylen   = (const int*)d_in[4];
    float* out = (float*)d_out;

    // Workspace carve-up
    char* ws = (char*)d_ws;
    float* negT = (float*)ws;                               size_t o = (size_t)BB * TX * TY * 4;   // 256 MB
    unsigned short* U  = (unsigned short*)(ws + o);         o += (size_t)BB * TY * K2 * 2;         // 48 MB
    unsigned short* Vm = (unsigned short*)(ws + o);         o += (size_t)BB * TX * K2 * 2;         // 6 MB
    float* cArr = (float*)(ws + o);                         o += (size_t)BB * TX * 4;
    unsigned* dir = (unsigned*)(ws + o);                    o += (size_t)BB * TX * (TY / 32) * 4;  // 8 MB
    int* idxA = (int*)(ws + o);                             o += (size_t)BB * TX * 4;

    // Output offsets (attn | w | m_p_exp | logs_p_exp)
    const size_t OFF_W = (size_t)BB * TY * TX;
    const size_t OFF_M = OFF_W + (size_t)BB * TX;
    const size_t OFF_L = OFF_M + (size_t)BB * DD * TY;

    prep_u<<<dim3(TY / 32, DD / 32, BB), dim3(32, 32), 0, stream>>>(z_p, U);
    prep_v<<<BB * TX, 256, 0, stream>>>(m_p, logs_p, Vm, cArr);

    const int nwaves = BB * (TX / 16) * (TY / 64);          // 65536 waves
    gemm_negcent<<<nwaves / 8, 256, 0, stream>>>(Vm, U, cArr, negT);

    dp_fwd<<<BB, 1024, 0, stream>>>(negT, ylen, dir);
    backtrack<<<1, 32, 0, stream>>>(dir, ylen, idxA);

    write_attn<<<(unsigned)(((size_t)BB * TY * TX) / 256), 256, 0, stream>>>(idxA, out);
    write_w<<<(BB * TX + 255) / 256, 256, 0, stream>>>(out + OFF_W);
    expand<<<(unsigned)(((size_t)BB * DD * TY) / 256), 256, 0, stream>>>(
        idxA, m_p, logs_p, out + OFF_M, out + OFF_L);
}
